// BDGNN_44418551775944
// MI455X (gfx1250) — matrix operand order repaired and checked by reading, not compiled, not run on hardware
//
#include <hip/hip_runtime.h>
#include <hip/hip_bf16.h>

typedef __attribute__((ext_vector_type(16))) _Float16 v16h;
typedef __attribute__((ext_vector_type(8)))  _Float16 v8h;
typedef __attribute__((ext_vector_type(8)))  float    v8f;

#define N_NODES 50000
#define N_EDGES 800000

// ---- workspace layout (bytes) ----
#define H_OFF    0u            // h   [50000 x 64] f32  = 12.8 MB
#define ACC_OFF  12800000u     // acc [50000 x 64] f32  = 12.8 MB
#define E_OFF    25600000u     // e   [800000 x 64] f32 = 204.8 MB
#define WP_OFF   230400000u    // packed f16 weight fragments (98 * 512 halves)

// ---- packed weight fragment offsets (units of 512 halves) ----
#define F_FE0 0
#define F_FE1 8
#define F_FE2 16
#define F_FV0 24
#define F_FV1 40
#define F_FV2 48
#define F_FA0 56
#define F_FA1 60
#define F_FB0 68
#define F_FB1 72
#define F_M10 80
#define F_M11 88
#define F_M12 96
#define F_TOTAL 98

#define STEP_HALVES (56*512)   // fe0..2 + fv0..2 fragments staged in LDS

// Cross-lane LDS RAW fence (also blocks compiler reordering of the ds ops)
#define LDS_FENCE() asm volatile("s_wait_dscnt 0" ::: "memory")

// ---------------- device helpers ----------------

__device__ __forceinline__ v8f v8f_zero() {
  v8f z;
#pragma unroll
  for (int i = 0; i < 8; ++i) z[i] = 0.0f;
  return z;
}

__device__ __forceinline__ v8f wmma_f16(v16h a, v16h b, v8f c) {
  return __builtin_amdgcn_wmma_f32_16x16x32_f16(false, a, false, b, (short)0, c,
                                                false, false);
}

__device__ __forceinline__ float softplus(float x) {
  return fmaxf(x, 0.0f) + __logf(1.0f + __expf(-fabsf(x)));
}

// A fragment (16x32 f16): halves 0..7 = K = kt*32 + half*8 + i,
//                         halves 8..15 = K = kt*32 + 16 + half*8 + i.
// Elementwise product of two fp32 rows (c2 = h[s]*h[r]).
__device__ __forceinline__ v16h frag_mul(const float* __restrict__ X,
                                         const float* __restrict__ Y,
                                         int kt, int half) {
  v16h r;
  int b0 = kt * 32 + half * 8, b1 = b0 + 16;
#pragma unroll
  for (int i = 0; i < 8; ++i) {
    r[i]     = (_Float16)(X[b0 + i] * Y[b0 + i]);
    r[i + 8] = (_Float16)(X[b1 + i] * Y[b1 + i]);
  }
  return r;
}

// A fragment from a single fp32 row.
__device__ __forceinline__ v16h frag_cvt(const float* __restrict__ X, int kt, int half) {
  v16h r;
  int b0 = kt * 32 + half * 8, b1 = b0 + 16;
#pragma unroll
  for (int i = 0; i < 8; ++i) {
    r[i]     = (_Float16)X[b0 + i];
    r[i + 8] = (_Float16)X[b1 + i];
  }
  return r;
}

// A fragment (kt==0) with K zero-padded beyond `valid` (for 16- and 8-wide inputs).
__device__ __forceinline__ v16h frag_cvt_pad(const float* __restrict__ X, int valid,
                                             int half) {
  v16h r;
  int b0 = half * 8, b1 = b0 + 16;
#pragma unroll
  for (int i = 0; i < 8; ++i) {
    int k0 = b0 + i, k1 = b1 + i;
    float x0 = X[k0 < valid ? k0 : 0];
    float x1 = X[k1 < valid ? k1 : 0];
    r[i]     = (k0 < valid) ? (_Float16)x0 : (_Float16)0.0f;
    r[i + 8] = (k1 < valid) ? (_Float16)x1 : (_Float16)0.0f;
  }
  return r;
}

// Load A fragment from a 16x64 f16 row-major LDS activation tile.
__device__ __forceinline__ v16h load_a_lds(const _Float16* __restrict__ A, int lane,
                                           int kt) {
  int row = lane & 15, half = lane >> 4;
  const _Float16* p = A + row * 64 + kt * 32 + half * 8;
  v8h lo = *(const v8h*)p;
  v8h hi = *(const v8h*)(p + 16);
  v16h r;
#pragma unroll
  for (int i = 0; i < 8; ++i) { r[i] = lo[i]; r[i + 8] = hi[i]; }
  return r;
}

// Load a pre-packed B fragment (lane-major, 16 halves per lane).
__device__ __forceinline__ v16h load_w(const _Float16* __restrict__ W, int kt, int nt,
                                       int ntc, int lane) {
  return *(const v16h*)(W + (((kt * ntc + nt) * 32 + lane) << 4));
}

// C-layout (VGPR j: row = 8*half + j; col = nt*16 + lane%16) -> LDS f16 tile,
// with bias + optional softplus.
__device__ __forceinline__ void store_act(_Float16* __restrict__ A, const v8f c[4],
                                          const float* __restrict__ bias, int lane,
                                          bool sp) {
  int col = lane & 15, rb = (lane >> 4) * 8;
#pragma unroll
  for (int nt = 0; nt < 4; ++nt) {
    float b = bias[nt * 16 + col];
#pragma unroll
    for (int j = 0; j < 8; ++j) {
      float v = c[nt][j] + b;
      if (sp) v = softplus(v);
      A[(rb + j) * 64 + nt * 16 + col] = (_Float16)v;
    }
  }
}

__device__ __forceinline__ void store_act_raw(_Float16* __restrict__ A, const v8f c[4],
                                              int lane) {
  int col = lane & 15, rb = (lane >> 4) * 8;
#pragma unroll
  for (int nt = 0; nt < 4; ++nt)
#pragma unroll
    for (int j = 0; j < 8; ++j)
      A[(rb + j) * 64 + nt * 16 + col] = (_Float16)c[nt][j];
}

// ---------------- weight packing ----------------
// Pack fp32 W[Ka x Na] (row-major, x@W convention) into f16 B fragments:
// dst[((kt*ntc+nt)*32 + lane)*16 + hh] = W[kt*32 + 8*(lane>>4) + (hh&7) + 16*(hh>>3)][nt*16 + (lane&15)]
__global__ void pack_w_kernel(const float* __restrict__ W, _Float16* __restrict__ dst,
                              int Ka, int Na, int ktc, int ntc) {
  int total = ktc * ntc * 512;
  for (int t = blockIdx.x * blockDim.x + threadIdx.x; t < total;
       t += gridDim.x * blockDim.x) {
    int f = t >> 9, w = t & 511, lane = w >> 4, hh = w & 15;
    int kt = f / ntc, nt = f % ntc;
    int K = kt * 32 + ((lane >> 4) * 8) + (hh & 7) + ((hh >> 3) * 16);
    int N = nt * 16 + (lane & 15);
    float v = (K < Ka && N < Na) ? W[K * Na + N] : 0.0f;
    dst[t] = (_Float16)v;
  }
}

// ---------------- node encoder (fa): x[N,16] -> h[N,64] ----------------
__global__ __launch_bounds__(256) void fa_kernel(const float* __restrict__ x,
                                                 float* __restrict__ h,
                                                 const _Float16* __restrict__ wp,
                                                 const float* __restrict__ b0,
                                                 const float* __restrict__ b1) {
  __shared__ __align__(32) _Float16 act[8][1024];
  int wave = threadIdx.x >> 5, lane = threadIdx.x & 31;
  int tile = blockIdx.x * 8 + wave;
  if (tile * 16 >= N_NODES) return;
  int nb = tile * 16;
  int row = lane & 15, half = lane >> 4, col = row, rb = half * 8;
  _Float16* A = act[wave];
  const _Float16* W0 = wp + F_FA0 * 512;
  const _Float16* W1 = wp + F_FA1 * 512;

  v8f c[4];
  {
    v16h a0 = frag_cvt_pad(x + (nb + row) * 16, 16, half);
#pragma unroll
    for (int nt = 0; nt < 4; ++nt) {
      v8f z = v8f_zero();
      z = wmma_f16(a0, load_w(W0, 0, nt, 4, lane), z);
      c[nt] = z;
    }
  }
  store_act(A, c, b0, lane, true);
  LDS_FENCE();
  {
    v16h t0 = load_a_lds(A, lane, 0), t1 = load_a_lds(A, lane, 1);
#pragma unroll
    for (int nt = 0; nt < 4; ++nt) {
      v8f z = v8f_zero();
      z = wmma_f16(t0, load_w(W1, 0, nt, 4, lane), z);
      z = wmma_f16(t1, load_w(W1, 1, nt, 4, lane), z);
      c[nt] = z;
    }
  }
#pragma unroll
  for (int nt = 0; nt < 4; ++nt) {
    float b = b1[nt * 16 + col];
#pragma unroll
    for (int j = 0; j < 8; ++j)
      h[(nb + rb + j) * 64 + nt * 16 + col] = c[nt][j] + b;
  }
}

// ---------------- edge encoder (fb): edge_attr[E,8] -> e[E,64] ----------------
__global__ __launch_bounds__(256) void fb_kernel(const float* __restrict__ ea,
                                                 float* __restrict__ e,
                                                 const _Float16* __restrict__ wp,
                                                 const float* __restrict__ b0,
                                                 const float* __restrict__ b1) {
  __shared__ __align__(32) _Float16 act[8][1024];
  int wave = threadIdx.x >> 5, lane = threadIdx.x & 31;
  int eb = (blockIdx.x * 8 + wave) * 16;          // exact: 800000 = 16*50000
  int row = lane & 15, half = lane >> 4, col = row, rb = half * 8;
  _Float16* A = act[wave];
  const _Float16* W0 = wp + F_FB0 * 512;
  const _Float16* W1 = wp + F_FB1 * 512;

  v8f c[4];
  {
    v16h a0 = frag_cvt_pad(ea + (eb + row) * 8, 8, half);
#pragma unroll
    for (int nt = 0; nt < 4; ++nt) {
      v8f z = v8f_zero();
      z = wmma_f16(a0, load_w(W0, 0, nt, 4, lane), z);
      c[nt] = z;
    }
  }
  store_act(A, c, b0, lane, true);
  LDS_FENCE();
  {
    v16h t0 = load_a_lds(A, lane, 0), t1 = load_a_lds(A, lane, 1);
#pragma unroll
    for (int nt = 0; nt < 4; ++nt) {
      v8f z = v8f_zero();
      z = wmma_f16(t0, load_w(W1, 0, nt, 4, lane), z);
      z = wmma_f16(t1, load_w(W1, 1, nt, 4, lane), z);
      c[nt] = z;
    }
  }
#pragma unroll
  for (int nt = 0; nt < 4; ++nt) {
    float b = b1[nt * 16 + col];
#pragma unroll
    for (int j = 0; j < 8; ++j)
      e[(eb + rb + j) * 64 + nt * 16 + col] = c[nt][j] + b;
  }
}

// ---------------- fused per-step edge kernel ----------------
struct EdgeParams {
  const float* h;
  float* e;
  float* acc;
  const int* s;
  const int* r;
  const _Float16* wp;   // packed step weights, fragment F_FE0 first
  const float *bfe0, *bfe1, *bfe2, *bfv0, *bfv1, *bfv2;
};

__global__ __launch_bounds__(256) void edge_step_kernel(EdgeParams p) {
  __shared__ __align__(32) _Float16 wlds[STEP_HALVES];  // 56 KB of f16 weights
  __shared__ __align__(32) _Float16 act[8][1024];       // 2 KB per wave
  {
    const uint4* src = (const uint4*)p.wp;
    uint4* dst = (uint4*)wlds;
    for (int i = threadIdx.x; i < STEP_HALVES / 8; i += 256) dst[i] = src[i];
  }
  __syncthreads();

  int wave = threadIdx.x >> 5, lane = threadIdx.x & 31;
  int eb = (blockIdx.x * 8 + wave) * 16;    // exact fit: 6250 blocks
  int row = lane & 15, half = lane >> 4, col = row, rb = half * 8;
  _Float16* A = act[wave];

  const _Float16* FE0 = wlds + (F_FE0 - F_FE0) * 512;
  const _Float16* FE1 = wlds + (F_FE1 - F_FE0) * 512;
  const _Float16* FE2 = wlds + (F_FE2 - F_FE0) * 512;
  const _Float16* FV0 = wlds + (F_FV0 - F_FE0) * 512;
  const _Float16* FV1 = wlds + (F_FV1 - F_FE0) * 512;
  const _Float16* FV2 = wlds + (F_FV2 - F_FE0) * 512;

  int sm = p.s[eb + row], rm = p.r[eb + row];
  const float* hs = p.h + sm * 64;
  const float* hr = p.h + rm * 64;

  v8f c[4];

  // ---- fe layer 0: A = h[s]*h[r] ----
  {
    v16h a0 = frag_mul(hs, hr, 0, half), a1 = frag_mul(hs, hr, 1, half);
#pragma unroll
    for (int nt = 0; nt < 4; ++nt) {
      v8f z = v8f_zero();
      z = wmma_f16(a0, load_w(FE0, 0, nt, 4, lane), z);
      z = wmma_f16(a1, load_w(FE0, 1, nt, 4, lane), z);
      c[nt] = z;
    }
  }
  store_act(A, c, p.bfe0, lane, true);
  LDS_FENCE();
  // ---- fe layer 1 ----
  {
    v16h t0 = load_a_lds(A, lane, 0), t1 = load_a_lds(A, lane, 1);
#pragma unroll
    for (int nt = 0; nt < 4; ++nt) {
      v8f z = v8f_zero();
      z = wmma_f16(t0, load_w(FE1, 0, nt, 4, lane), z);
      z = wmma_f16(t1, load_w(FE1, 1, nt, 4, lane), z);
      c[nt] = z;
    }
  }
  store_act(A, c, p.bfe1, lane, true);
  LDS_FENCE();
  // ---- fe layer 2 (linear) + residual into e ----
  {
    v16h t0 = load_a_lds(A, lane, 0), t1 = load_a_lds(A, lane, 1);
#pragma unroll
    for (int nt = 0; nt < 4; ++nt) {
      v8f z = v8f_zero();
      z = wmma_f16(t0, load_w(FE2, 0, nt, 4, lane), z);
      z = wmma_f16(t1, load_w(FE2, 1, nt, 4, lane), z);
      c[nt] = z;
    }
  }
#pragma unroll
  for (int nt = 0; nt < 4; ++nt) {
    float b = p.bfe2[nt * 16 + col];
#pragma unroll
    for (int j = 0; j < 8; ++j) {
      float* ep = p.e + (eb + rb + j) * 64 + nt * 16 + col;
      float v = c[nt][j] + b + *ep;   // e' = fe(...) + e
      *ep = v;
      c[nt][j] = v;
    }
  }
  store_act_raw(A, c, lane);          // e' f16 for fv K=64..127
  LDS_FENCE();
  // ---- fv layer 0: A = concat(h[r], e') ----
  {
    v16h a0 = frag_cvt(hr, 0, half), a1 = frag_cvt(hr, 1, half);
    v16h a2 = load_a_lds(A, lane, 0), a3 = load_a_lds(A, lane, 1);
#pragma unroll
    for (int nt = 0; nt < 4; ++nt) {
      v8f z = v8f_zero();
      z = wmma_f16(a0, load_w(FV0, 0, nt, 4, lane), z);
      z = wmma_f16(a1, load_w(FV0, 1, nt, 4, lane), z);
      z = wmma_f16(a2, load_w(FV0, 2, nt, 4, lane), z);
      z = wmma_f16(a3, load_w(FV0, 3, nt, 4, lane), z);
      c[nt] = z;
    }
  }
  store_act(A, c, p.bfv0, lane, true);
  LDS_FENCE();
  // ---- fv layer 1 ----
  {
    v16h t0 = load_a_lds(A, lane, 0), t1 = load_a_lds(A, lane, 1);
#pragma unroll
    for (int nt = 0; nt < 4; ++nt) {
      v8f z = v8f_zero();
      z = wmma_f16(t0, load_w(FV1, 0, nt, 4, lane), z);
      z = wmma_f16(t1, load_w(FV1, 1, nt, 4, lane), z);
      c[nt] = z;
    }
  }
  store_act(A, c, p.bfv1, lane, true);
  LDS_FENCE();
  // ---- fv layer 2 (linear) + scatter-add ----
  {
    v16h t0 = load_a_lds(A, lane, 0), t1 = load_a_lds(A, lane, 1);
#pragma unroll
    for (int nt = 0; nt < 4; ++nt) {
      v8f z = v8f_zero();
      z = wmma_f16(t0, load_w(FV2, 0, nt, 4, lane), z);
      z = wmma_f16(t1, load_w(FV2, 1, nt, 4, lane), z);
      c[nt] = z;
    }
  }
  int rr[8];
#pragma unroll
  for (int j = 0; j < 8; ++j) rr[j] = p.r[eb + rb + j];
#pragma unroll
  for (int nt = 0; nt < 4; ++nt) {
    float b = p.bfv2[nt * 16 + col];
#pragma unroll
    for (int j = 0; j < 8; ++j)
      atomicAdd(p.acc + rr[j] * 64 + nt * 16 + col, c[nt][j] + b);
  }
}

// ---------------- node residual update: h += acc; acc = 0 ----------------
__global__ void update_kernel(float* __restrict__ h, float* __restrict__ acc) {
  int i = blockIdx.x * blockDim.x + threadIdx.x;  // exact: 12500*256 = N*64
  h[i] += acc[i];
  acc[i] = 0.0f;
}

// ---------------- force head (mlp1): h[N,64] -> out[N,3] ----------------
__global__ __launch_bounds__(256) void force_kernel(const float* __restrict__ h,
                                                    float* __restrict__ out,
                                                    const _Float16* __restrict__ wp,
                                                    const float* __restrict__ b0,
                                                    const float* __restrict__ b1,
                                                    const float* __restrict__ b2) {
  __shared__ __align__(32) _Float16 act[8][1024];
  int wave = threadIdx.x >> 5, lane = threadIdx.x & 31;
  int tile = blockIdx.x * 8 + wave;
  if (tile * 16 >= N_NODES) return;
  int nb = tile * 16;
  int row = lane & 15, half = lane >> 4, col = row, rb = half * 8;
  _Float16* A = act[wave];
  const _Float16* W0 = wp + F_M10 * 512;
  const _Float16* W1 = wp + F_M11 * 512;
  const _Float16* W2 = wp + F_M12 * 512;

  v8f c[4];
  {
    const float* hrow = h + (nb + row) * 64;
    v16h a0 = frag_cvt(hrow, 0, half), a1 = frag_cvt(hrow, 1, half);
#pragma unroll
    for (int nt = 0; nt < 4; ++nt) {
      v8f z = v8f_zero();
      z = wmma_f16(a0, load_w(W0, 0, nt, 4, lane), z);
      z = wmma_f16(a1, load_w(W0, 1, nt, 4, lane), z);
      c[nt] = z;
    }
  }
  store_act(A, c, b0, lane, true);
  LDS_FENCE();
  {
    v16h t0 = load_a_lds(A, lane, 0), t1 = load_a_lds(A, lane, 1);
#pragma unroll
    for (int nt = 0; nt < 4; ++nt) {
      v8f z = v8f_zero();
      z = wmma_f16(t0, load_w(W1, 0, nt, 4, lane), z);
      z = wmma_f16(t1, load_w(W1, 1, nt, 4, lane), z);
      c[nt] = z;
    }
  }
  store_act(A, c, b1, lane, true);
  LDS_FENCE();
  {
    v16h t0 = load_a_lds(A, lane, 0), t1 = load_a_lds(A, lane, 1);
    v8f z = v8f_zero();
    z = wmma_f16(t0, load_w(W2, 0, 0, 1, lane), z);
    z = wmma_f16(t1, load_w(W2, 1, 0, 1, lane), z);
    if (col < 3) {
      float b = b2[col];
#pragma unroll
      for (int j = 0; j < 8; ++j) out[(nb + rb + j) * 3 + col] = z[j] + b;
    }
  }
}

// ---------------- gamma head (mlp2): type_ohe[N,5] -> out[N,1] ----------------
__global__ void gamma_kernel(const float* __restrict__ t, const float* __restrict__ W0,
                             const float* __restrict__ b0, const float* __restrict__ W1,
                             const float* __restrict__ b1, const float* __restrict__ W2,
                             const float* __restrict__ b2, float* __restrict__ out) {
  int i = blockIdx.x * blockDim.x + threadIdx.x;
  if (i >= N_NODES) return;
  float in[5];
#pragma unroll
  for (int k = 0; k < 5; ++k) in[k] = t[i * 5 + k];
  float h1[10];
#pragma unroll
  for (int o = 0; o < 10; ++o) {
    float s = b0[o];
#pragma unroll
    for (int k = 0; k < 5; ++k) s += in[k] * W0[k * 10 + o];
    h1[o] = softplus(s);
  }
  float h2[5];
#pragma unroll
  for (int o = 0; o < 5; ++o) {
    float s = b1[o];
#pragma unroll
    for (int k = 0; k < 10; ++k) s += h1[k] * W1[k * 5 + o];
    h2[o] = softplus(s);
  }
  float s = b2[0];
#pragma unroll
  for (int k = 0; k < 5; ++k) s += h2[k] * W2[k];
  out[i] = softplus(s);
}

// ---------------- host launcher ----------------
extern "C" void kernel_launch(void* const* d_in, const int* in_sizes, int n_in,
                              void* d_out, int out_size, void* d_ws, size_t ws_size,
                              hipStream_t stream) {
  const float* x     = (const float*)d_in[0];
  const int*   eidx  = (const int*)d_in[1];
  const float* eattr = (const float*)d_in[2];
  const float* tohe  = (const float*)d_in[3];
  // params (tree-flattened: fa, fb, fe, fv, mlp1, mlp2; each layer W then b)
  const float* faW0 = (const float*)d_in[4];  const float* fab0 = (const float*)d_in[5];
  const float* faW1 = (const float*)d_in[6];  const float* fab1 = (const float*)d_in[7];
  const float* fbW0 = (const float*)d_in[8];  const float* fbb0 = (const float*)d_in[9];
  const float* fbW1 = (const float*)d_in[10]; const float* fbb1 = (const float*)d_in[11];
  const float* feW0 = (const float*)d_in[12]; const float* feb0 = (const float*)d_in[13];
  const float* feW1 = (const float*)d_in[14]; const float* feb1 = (const float*)d_in[15];
  const float* feW2 = (const float*)d_in[16]; const float* feb2 = (const float*)d_in[17];
  const float* fvW0 = (const float*)d_in[18]; const float* fvb0 = (const float*)d_in[19];
  const float* fvW1 = (const float*)d_in[20]; const float* fvb1 = (const float*)d_in[21];
  const float* fvW2 = (const float*)d_in[22]; const float* fvb2 = (const float*)d_in[23];
  const float* m1W0 = (const float*)d_in[24]; const float* m1b0 = (const float*)d_in[25];
  const float* m1W1 = (const float*)d_in[26]; const float* m1b1 = (const float*)d_in[27];
  const float* m1W2 = (const float*)d_in[28]; const float* m1b2 = (const float*)d_in[29];
  const float* m2W0 = (const float*)d_in[30]; const float* m2b0 = (const float*)d_in[31];
  const float* m2W1 = (const float*)d_in[32]; const float* m2b1 = (const float*)d_in[33];
  const float* m2W2 = (const float*)d_in[34]; const float* m2b2 = (const float*)d_in[35];

  char* ws = (char*)d_ws;
  float* h   = (float*)(ws + H_OFF);
  float* acc = (float*)(ws + ACC_OFF);
  float* e   = (float*)(ws + E_OFF);
  _Float16* wp = (_Float16*)(ws + WP_OFF);

  // pack all weight matrices into WMMA B-fragment order (f16)
  struct PD { const float* W; int Ka, Na, ktc, ntc, foff; };
  const PD pd[13] = {
      {feW0, 64, 64, 2, 4, F_FE0},  {feW1, 64, 64, 2, 4, F_FE1},
      {feW2, 64, 64, 2, 4, F_FE2},  {fvW0, 128, 64, 4, 4, F_FV0},
      {fvW1, 64, 64, 2, 4, F_FV1},  {fvW2, 64, 64, 2, 4, F_FV2},
      {faW0, 16, 64, 1, 4, F_FA0},  {faW1, 64, 64, 2, 4, F_FA1},
      {fbW0, 8, 64, 1, 4, F_FB0},   {fbW1, 64, 64, 2, 4, F_FB1},
      {m1W0, 64, 64, 2, 4, F_M10},  {m1W1, 64, 64, 2, 4, F_M11},
      {m1W2, 64, 3, 2, 1, F_M12},
  };
  for (int i = 0; i < 13; ++i) {
    int total = pd[i].ktc * pd[i].ntc * 512;
    pack_w_kernel<<<(total + 255) / 256, 256, 0, stream>>>(
        pd[i].W, wp + pd[i].foff * 512, pd[i].Ka, pd[i].Na, pd[i].ktc, pd[i].ntc);
  }

  hipMemsetAsync(acc, 0, (size_t)N_NODES * 64 * sizeof(float), stream);

  // encoders
  fa_kernel<<<391, 256, 0, stream>>>(x, h, wp, fab0, fab1);       // 3125 tiles
  fb_kernel<<<6250, 256, 0, stream>>>(eattr, e, wp, fbb0, fbb1);  // 50000 tiles

  // message-passing steps
  EdgeParams ep;
  ep.h = h; ep.e = e; ep.acc = acc;
  ep.s = eidx; ep.r = eidx + N_EDGES;
  ep.wp = wp;  // step weights start at fragment 0
  ep.bfe0 = feb0; ep.bfe1 = feb1; ep.bfe2 = feb2;
  ep.bfv0 = fvb0; ep.bfv1 = fvb1; ep.bfv2 = fvb2;
  for (int step = 0; step < 5; ++step) {
    edge_step_kernel<<<6250, 256, 0, stream>>>(ep);
    update_kernel<<<(N_NODES * 64) / 256, 256, 0, stream>>>(h, acc);
  }

  // heads
  float* out = (float*)d_out;
  force_kernel<<<391, 256, 0, stream>>>(h, out, wp, m1b0, m1b1, m1b2);
  gamma_kernel<<<(N_NODES + 255) / 256, 256, 0, stream>>>(
      tohe, m2W0, m2b0, m2W1, m2b1, m2W2, m2b2, out + N_NODES * 3);
}
